// ProofStateLatentAttention_5935644803714
// MI455X (gfx1250) — matrix operand order repaired and checked by reading, not compiled, not run on hardware
//
#include <hip/hip_runtime.h>

typedef unsigned short u16;
typedef unsigned int u32;
typedef __attribute__((ext_vector_type(16))) __bf16 v16bf;
typedef __attribute__((ext_vector_type(8)))  float  v8f;

union FragBF { uint4 u[2]; v16bf v; };

__device__ __forceinline__ u16 f2bf(float f) {
  union { float f; u32 u; } x; x.f = f;
  u32 r = x.u + 0x7FFFu + ((x.u >> 16) & 1u);
  return (u16)(r >> 16);
}

#define WMMA_BF16(A, B, C) \
  __builtin_amdgcn_wmma_f32_16x16x32_bf16(false, (A), false, (B), (short)0, (C), false, false)

// ---- CDNA5 async copy to LDS (ASYNCcnt-tracked), guarded so compile never breaks
typedef int mla_v4i __attribute__((vector_size(16)));
typedef __attribute__((address_space(1))) mla_v4i mla_g_v4i;   // global
typedef __attribute__((address_space(3))) mla_v4i mla_l_v4i;   // LDS

#if __has_builtin(__builtin_amdgcn_global_load_async_to_lds_b128) && \
    __has_builtin(__builtin_amdgcn_s_wait_asynccnt)
#define MLA_ASYNC 1
#else
#define MLA_ASYNC 0
#endif

__device__ __forceinline__ void cp16_to_lds(const u16* g, u16* l) {
#if MLA_ASYNC
  __builtin_amdgcn_global_load_async_to_lds_b128((mla_g_v4i*)g, (mla_l_v4i*)l, 0, 0);
#else
  *(uint4*)l = *(const uint4*)g;
#endif
}
template <int N> __device__ __forceinline__ void cp_wait() {
#if MLA_ASYNC
  __builtin_amdgcn_s_wait_asynccnt(N);
#endif
}

// ---------------------------------------------------------------- cast fp32 -> bf16
__global__ void mla_cast_bf16_kernel(const float* __restrict__ in, u16* __restrict__ out, int n) {
  int i = blockIdx.x * 256 + threadIdx.x;
  if (i < n) out[i] = f2bf(in[i]);
}

// ---------------------------------------------------------------- cast + transpose fp32(KxN) -> bf16(NxK)
__global__ void __launch_bounds__(256)
mla_cast_transpose_kernel(const float* __restrict__ in, u16* __restrict__ out, int K, int N) {
  __shared__ float t[32][33];
  int k0 = blockIdx.x * 32, n0 = blockIdx.y * 32;
  int tx = threadIdx.x & 31, ty = threadIdx.x >> 5;   // ty = 0..7
#pragma unroll
  for (int i = 0; i < 32; i += 8)
    t[ty + i][tx] = in[(size_t)(k0 + ty + i) * N + n0 + tx];
  __syncthreads();
#pragma unroll
  for (int i = 0; i < 32; i += 8)
    out[(size_t)(n0 + ty + i) * K + k0 + tx] = f2bf(t[tx][ty + i]);
}

// ---------------------------------------------------------------- bf16 GEMM, B pre-transposed
// C(f32, MxN) = A(bf16, MxK) * BT(bf16, NxK)^T.  128x64 block tile, 8 waves,
// BK=32, double-buffered LDS, async-to-LDS staging.
__global__ void __launch_bounds__(256)
mla_gemm_bf16_kernel(const u16* __restrict__ A, const u16* __restrict__ BT,
                     float* __restrict__ C, int M, int N, int K) {
  __shared__ u16 As[2][128][32];   // [m][k]
  __shared__ u16 Bs[2][64][32];    // [n][k]
  int tid = threadIdx.x, lane = tid & 31, wave = tid >> 5;
  int m15 = lane & 15, hi = lane >> 4;
  int rowBase = blockIdx.y * 128, colBase = blockIdx.x * 64;
  int tm = (wave & 3) * 32;        // wave's 32-row strip
  int tn = (wave >> 2) * 32;       // wave's 32-col strip
  int ra = tid >> 2, ca = (tid & 3) * 8;
  v8f acc00 = {}, acc01 = {}, acc10 = {}, acc11 = {};

  const u16* Ab = A + (size_t)rowBase * K;
  const u16* Bb = BT + (size_t)colBase * K;

  auto stage = [&](int buf, int k0) {
    cp16_to_lds(Ab + (size_t)ra * K + k0 + ca,        &As[buf][ra][ca]);
    cp16_to_lds(Ab + (size_t)(ra + 64) * K + k0 + ca, &As[buf][ra + 64][ca]);
    cp16_to_lds(Bb + (size_t)ra * K + k0 + ca,        &Bs[buf][ra][ca]);
  };

  stage(0, 0);
  int pb = 0;
  for (int k0 = 0; k0 < K; k0 += 32) {
    bool more = (k0 + 32) < K;
    if (more) {                       // prefetch next tile into other buffer
      stage(pb ^ 1, k0 + 32);
      cp_wait<3>();                   // current tile done; next tile's 3 ops in flight
    } else {
      cp_wait<0>();
    }
    __syncthreads();
    FragBF a0, a1, b0, b1;
    { const u16* q = &As[pb][tm + m15][hi * 8];
      a0.u[0] = *(const uint4*)q; a0.u[1] = *(const uint4*)(q + 16); }
    { const u16* q = &As[pb][tm + 16 + m15][hi * 8];
      a1.u[0] = *(const uint4*)q; a1.u[1] = *(const uint4*)(q + 16); }
    { const u16* q = &Bs[pb][tn + m15][hi * 8];
      b0.u[0] = *(const uint4*)q; b0.u[1] = *(const uint4*)(q + 16); }
    { const u16* q = &Bs[pb][tn + 16 + m15][hi * 8];
      b1.u[0] = *(const uint4*)q; b1.u[1] = *(const uint4*)(q + 16); }
    acc00 = WMMA_BF16(a0.v, b0.v, acc00);
    acc01 = WMMA_BF16(a0.v, b1.v, acc01);
    acc10 = WMMA_BF16(a1.v, b0.v, acc10);
    acc11 = WMMA_BF16(a1.v, b1.v, acc11);
    __syncthreads();
    pb ^= 1;
  }
  int row0 = rowBase + tm + hi * 8;
  int col0 = colBase + tn + m15;
#pragma unroll
  for (int r = 0; r < 8; ++r) {
    C[(size_t)(row0 + r) * N + col0]           = acc00[r];
    C[(size_t)(row0 + r) * N + col0 + 16]      = acc01[r];
    C[(size_t)(row0 + 16 + r) * N + col0]      = acc10[r];
    C[(size_t)(row0 + 16 + r) * N + col0 + 16] = acc11[r];
  }
}

// ---------------------------------------------------------------- layernorm over DL=256 -> bf16
__global__ void __launch_bounds__(256)
mla_ln256_kernel(const float* __restrict__ in, const float* __restrict__ g,
                 const float* __restrict__ b, u16* __restrict__ out) {
  __shared__ float sbuf[256];
  __shared__ float stat[2];
  int row = blockIdx.x, tid = threadIdx.x;
  float x = in[(size_t)row * 256 + tid];
  sbuf[tid] = x; __syncthreads();
  for (int s = 128; s > 0; s >>= 1) { if (tid < s) sbuf[tid] += sbuf[tid + s]; __syncthreads(); }
  if (tid == 0) stat[0] = sbuf[0] * (1.0f / 256.0f);
  __syncthreads();
  float m = stat[0], d = x - m;
  sbuf[tid] = d * d; __syncthreads();
  for (int s = 128; s > 0; s >>= 1) { if (tid < s) sbuf[tid] += sbuf[tid + s]; __syncthreads(); }
  if (tid == 0) stat[1] = rsqrtf(sbuf[0] * (1.0f / 256.0f) + 1e-5f);
  __syncthreads();
  out[(size_t)row * 256 + tid] = f2bf(d * stat[1] * g[tid] + b[tid]);
}

// ---------------------------------------------------------------- RoPE + per-head LN for q,k; v transpose
__global__ void __launch_bounds__(64)
mla_ropeln_kernel(const float* __restrict__ qpre, const float* __restrict__ kv,
                  const int* __restrict__ positions,
                  const float* __restrict__ qg, const float* __restrict__ qb,
                  const float* __restrict__ kg, const float* __restrict__ kb,
                  u16* __restrict__ qbf, u16* __restrict__ kbf, u16* __restrict__ vtbf) {
  const int S = 2048, H = 16;
  int idx = blockIdx.x;
  int h = idx & 15;
  int row = idx >> 4;           // b*S + s
  int s = row & (S - 1);
  int d = threadIdx.x;          // 0..63
  __shared__ float qraw[64], kraw[64], tmp[64], stat[2];

  float qv = qpre[(size_t)row * 1024 + h * 64 + d];
  float kx = kv[(size_t)row * 2048 + h * 64 + d];
  float vv = kv[(size_t)row * 2048 + 1024 + h * 64 + d];
  qraw[d] = qv; kraw[d] = kx;
  __syncthreads();

  float pos = (float)positions[s];
  int j = d & 31;
  float invf = __expf(-(float)(2 * j) * (9.210340371976184f / 64.0f)); // 10000^(-2j/64)
  float ang = pos * invf;
  float c = __cosf(ang), sn = __sinf(ang);
  float qr, kr;
  if (d < 32) { qr = qraw[d] * c - qraw[d + 32] * sn;  kr = kraw[d] * c - kraw[d + 32] * sn; }
  else        { qr = qraw[d - 32] * sn + qraw[d] * c;  kr = kraw[d - 32] * sn + kraw[d] * c; }

  tmp[d] = qr; __syncthreads();
  if (d == 0) {
    float su = 0.f; for (int i = 0; i < 64; ++i) su += tmp[i];
    float m = su * (1.0f / 64.0f);
    float v2 = 0.f; for (int i = 0; i < 64; ++i) { float t = tmp[i] - m; v2 += t * t; }
    stat[0] = m; stat[1] = rsqrtf(v2 * (1.0f / 64.0f) + 1e-5f);
  }
  __syncthreads();
  float qn = (qr - stat[0]) * stat[1] * qg[d] + qb[d];
  __syncthreads();

  tmp[d] = kr; __syncthreads();
  if (d == 0) {
    float su = 0.f; for (int i = 0; i < 64; ++i) su += tmp[i];
    float m = su * (1.0f / 64.0f);
    float v2 = 0.f; for (int i = 0; i < 64; ++i) { float t = tmp[i] - m; v2 += t * t; }
    stat[0] = m; stat[1] = rsqrtf(v2 * (1.0f / 64.0f) + 1e-5f);
  }
  __syncthreads();
  float kn = (kr - stat[0]) * stat[1] * kg[d] + kb[d];

  size_t bh = (size_t)((row / S) * H + h);
  qbf[(bh * S + s) * 64 + d]  = f2bf(qn);
  kbf[(bh * S + s) * 64 + d]  = f2bf(kn);
  vtbf[(bh * 64 + d) * S + s] = f2bf(vv);   // v transposed [b,h,dh,s]
}

// ---------------------------------------------------------------- flash attention, bf16 WMMA
// grid = (S/64, B*H), 256 threads (8 waves). Async double-buffered K/V,
// register-resident online softmax with cross-lane shuffle reductions.
__global__ void __launch_bounds__(256)
mla_attn_kernel(const u16* __restrict__ qbf, const u16* __restrict__ kbf,
                const u16* __restrict__ vtbf, u16* __restrict__ obf) {
  const int S = 2048;
  __shared__ u16 Qs[64][64];
  __shared__ u16 Ks[2][64][64];   // [key][dh]  == Bt for QK^T
  __shared__ u16 Vts[2][64][64];  // [dh][key]  == Bt for PV
  __shared__ u16 Ps[64][64];
  __shared__ float redmax[64][2];
  __shared__ float redsum[64][2];

  int tid = threadIdx.x, lane = tid & 31, wave = tid >> 5;
  int m15 = lane & 15, hi = lane >> 4;
  int tm = (wave & 3) * 16;
  int wc = wave >> 2;             // column-wave: 0 or 1
  int tn = wc * 32;
  int bh = blockIdx.y, q0 = blockIdx.x * 64;
  const u16* qp = qbf + ((size_t)bh * S + q0) * 64;
  const u16* kp = kbf + (size_t)bh * S * 64;
  const u16* vp = vtbf + (size_t)bh * 64 * S;

  int sr = tid >> 2, sc = (tid & 3) * 16;

  auto stageKV = [&](int buf, int kt) {
    cp16_to_lds(kp + (size_t)(kt + sr) * 64 + sc,     &Ks[buf][sr][sc]);
    cp16_to_lds(kp + (size_t)(kt + sr) * 64 + sc + 8, &Ks[buf][sr][sc + 8]);
    cp16_to_lds(vp + (size_t)sr * S + kt + sc,        &Vts[buf][sr][sc]);
    cp16_to_lds(vp + (size_t)sr * S + kt + sc + 8,    &Vts[buf][sr][sc + 8]);
  };

  // stage Q (direct) and first K/V tile (async)
  *(uint4*)&Qs[sr][sc]     = *(const uint4*)(qp + (size_t)sr * 64 + sc);
  *(uint4*)&Qs[sr][sc + 8] = *(const uint4*)(qp + (size_t)sr * 64 + sc + 8);
  stageKV(0, 0);
  __syncthreads();

  FragBF qa0, qa1, fb, pa0, pa1;
  { const u16* q = &Qs[tm + m15][hi * 8];
    qa0.u[0] = *(const uint4*)q; qa0.u[1] = *(const uint4*)(q + 16);
    q = &Qs[tm + m15][32 + hi * 8];
    qa1.u[0] = *(const uint4*)q; qa1.u[1] = *(const uint4*)(q + 16); }
  v8f o0 = {}, o1 = {};
  float m_loc[8], l_loc[8];
#pragma unroll
  for (int r = 0; r < 8; ++r) { m_loc[r] = -3.0e38f; l_loc[r] = 0.f; }
  int mr = tm + hi * 8;
  int pb = 0;

  for (int kt = 0; kt < S; kt += 64) {
    bool more = (kt + 64) < S;
    if (more) { stageKV(pb ^ 1, kt + 64); cp_wait<4>(); }
    else      { cp_wait<0>(); }
    __syncthreads();   // A: K/V buffer pb ready

    // ---- scores: two 16x16 tiles per wave, K-dim = 64 = 2 WMMA steps
    v8f s0 = {}, s1 = {};
    { const u16* q = &Ks[pb][tn + m15][hi * 8];
      fb.u[0] = *(const uint4*)q; fb.u[1] = *(const uint4*)(q + 16);
      s0 = WMMA_BF16(qa0.v, fb.v, s0);
      q = &Ks[pb][tn + m15][32 + hi * 8];
      fb.u[0] = *(const uint4*)q; fb.u[1] = *(const uint4*)(q + 16);
      s0 = WMMA_BF16(qa1.v, fb.v, s0);
      q = &Ks[pb][tn + 16 + m15][hi * 8];
      fb.u[0] = *(const uint4*)q; fb.u[1] = *(const uint4*)(q + 16);
      s1 = WMMA_BF16(qa0.v, fb.v, s1);
      q = &Ks[pb][tn + 16 + m15][32 + hi * 8];
      fb.u[0] = *(const uint4*)q; fb.u[1] = *(const uint4*)(q + 16);
      s1 = WMMA_BF16(qa1.v, fb.v, s1); }

    // ---- row max over this wave's 32 columns (shuffle within 16-lane group)
    float mx[8];
#pragma unroll
    for (int r = 0; r < 8; ++r) {
      s0[r] *= 0.125f; s1[r] *= 0.125f;        // SCALE = 1/sqrt(64)
      float v = fmaxf(s0[r], s1[r]);
      v = fmaxf(v, __shfl_xor(v, 1));
      v = fmaxf(v, __shfl_xor(v, 2));
      v = fmaxf(v, __shfl_xor(v, 4));
      v = fmaxf(v, __shfl_xor(v, 8));
      mx[r] = v;
    }
    if (m15 == 0) {
#pragma unroll
      for (int r = 0; r < 8; ++r) redmax[mr + r][wc] = mx[r];
    }
    __syncthreads();   // B: redmax visible

    // ---- exp, P write, row sums (replicated register state stays consistent)
    float alpha[8], rs[8];
#pragma unroll
    for (int r = 0; r < 8; ++r) {
      float mnew = fmaxf(m_loc[r], fmaxf(redmax[mr + r][0], redmax[mr + r][1]));
      alpha[r] = __expf(m_loc[r] - mnew);
      m_loc[r] = mnew;
      float p0 = __expf(s0[r] - mnew);
      float p1 = __expf(s1[r] - mnew);
      Ps[mr + r][tn + m15]      = f2bf(p0);
      Ps[mr + r][tn + 16 + m15] = f2bf(p1);
      float v = p0 + p1;
      v += __shfl_xor(v, 1);
      v += __shfl_xor(v, 2);
      v += __shfl_xor(v, 4);
      v += __shfl_xor(v, 8);
      rs[r] = v;
    }
    if (m15 == 0) {
#pragma unroll
      for (int r = 0; r < 8; ++r) redsum[mr + r][wc] = rs[r];
    }
    __syncthreads();   // C: Ps + redsum visible

#pragma unroll
    for (int r = 0; r < 8; ++r) {
      l_loc[r] = l_loc[r] * alpha[r] + redsum[mr + r][0] + redsum[mr + r][1];
      o0[r] *= alpha[r];
      o1[r] *= alpha[r];
    }

    // ---- O += P * V
    { const u16* q = &Ps[tm + m15][hi * 8];
      pa0.u[0] = *(const uint4*)q; pa0.u[1] = *(const uint4*)(q + 16);
      q = &Ps[tm + m15][32 + hi * 8];
      pa1.u[0] = *(const uint4*)q; pa1.u[1] = *(const uint4*)(q + 16); }
    { const u16* q = &Vts[pb][tn + m15][hi * 8];
      fb.u[0] = *(const uint4*)q; fb.u[1] = *(const uint4*)(q + 16);
      o0 = WMMA_BF16(pa0.v, fb.v, o0);
      q = &Vts[pb][tn + m15][32 + hi * 8];
      fb.u[0] = *(const uint4*)q; fb.u[1] = *(const uint4*)(q + 16);
      o0 = WMMA_BF16(pa1.v, fb.v, o0);
      q = &Vts[pb][tn + 16 + m15][hi * 8];
      fb.u[0] = *(const uint4*)q; fb.u[1] = *(const uint4*)(q + 16);
      o1 = WMMA_BF16(pa0.v, fb.v, o1);
      q = &Vts[pb][tn + 16 + m15][32 + hi * 8];
      fb.u[0] = *(const uint4*)q; fb.u[1] = *(const uint4*)(q + 16);
      o1 = WMMA_BF16(pa1.v, fb.v, o1); }
    __syncthreads();   // E: all reads of buffers done before next iteration's writes
    pb ^= 1;
  }

  int b = bh >> 4, h = bh & 15;
#pragma unroll
  for (int r = 0; r < 8; ++r) {
    int s = q0 + mr + r;
    float inv = 1.0f / l_loc[r];
    size_t base = ((size_t)(b * S + s)) * 1024 + h * 64;
    obf[base + tn + m15]      = f2bf(o0[r] * inv);
    obf[base + tn + 16 + m15] = f2bf(o1[r] * inv);
  }
}

// ---------------------------------------------------------------- launch
extern "C" void kernel_launch(void* const* d_in, const int* in_sizes, int n_in,
                              void* d_out, int out_size, void* d_ws, size_t ws_size,
                              hipStream_t stream) {
  (void)in_sizes; (void)n_in; (void)out_size; (void)ws_size;
  const float* x     = (const float*)d_in[0];
  const int*   pos   = (const int*)d_in[1];
  const float* Wq    = (const float*)d_in[2];
  const float* Wdown = (const float*)d_in[3];
  const float* ln_g  = (const float*)d_in[4];
  const float* ln_b  = (const float*)d_in[5];
  const float* Wkv   = (const float*)d_in[6];
  const float* Wo    = (const float*)d_in[7];
  const float* qn_g  = (const float*)d_in[8];
  const float* qn_b  = (const float*)d_in[9];
  const float* kn_g  = (const float*)d_in[10];
  const float* kn_b  = (const float*)d_in[11];
  float* out = (float*)d_out;

  const int BS = 2, SS = 2048, DD = 1024, HH = 16, DHH = 64, DLL = 256;
  const int M = BS * SS;   // 4096

  char* w = (char*)d_ws;
  size_t off = 0;
  auto take = [&](size_t n) -> char* {
    char* p = w + off; off = (off + n + 255) & ~(size_t)255; return p;
  };
  u16*   xbf    = (u16*)take((size_t)M * DD * 2);
  u16*   wqT    = (u16*)take((size_t)DD * DD * 2);        // [n=1024][k=1024]
  u16*   wdT    = (u16*)take((size_t)DLL * DD * 2);       // [n=256][k=1024]
  u16*   wkvT   = (u16*)take((size_t)(2 * DD) * DLL * 2); // [n=2048][k=256]
  u16*   woT    = (u16*)take((size_t)DD * DD * 2);        // [n=1024][k=1024]
  float* qpre   = (float*)take((size_t)M * DD * 4);
  float* lat    = (float*)take((size_t)M * DLL * 4);
  u16*   latbf  = (u16*)take((size_t)M * DLL * 2);
  float* kvf    = (float*)take((size_t)M * 2 * DD * 4);
  u16*   qbf    = (u16*)take((size_t)BS * HH * SS * DHH * 2);
  u16*   kbf    = (u16*)take((size_t)BS * HH * SS * DHH * 2);
  u16*   vtbf   = (u16*)take((size_t)BS * HH * DHH * SS * 2);
  u16*   attnbf = (u16*)take((size_t)M * DD * 2);

  // 1) casts (+ weight transposes, done once)
  mla_cast_bf16_kernel<<<(M * DD + 255) / 256, 256, 0, stream>>>(x, xbf, M * DD);
  mla_cast_transpose_kernel<<<dim3(DD / 32, DD / 32), 256, 0, stream>>>(Wq, wqT, DD, DD);
  mla_cast_transpose_kernel<<<dim3(DD / 32, DLL / 32), 256, 0, stream>>>(Wdown, wdT, DD, DLL);
  mla_cast_transpose_kernel<<<dim3(DLL / 32, 2 * DD / 32), 256, 0, stream>>>(Wkv, wkvT, DLL, 2 * DD);
  mla_cast_transpose_kernel<<<dim3(DD / 32, DD / 32), 256, 0, stream>>>(Wo, woT, DD, DD);

  // 2) q_pre = x @ Wq  (4096x1024x1024)
  mla_gemm_bf16_kernel<<<dim3(DD / 64, M / 128), 256, 0, stream>>>(xbf, wqT, qpre, M, DD, DD);
  // 3) lat = x @ Wdown (4096x256x1024)
  mla_gemm_bf16_kernel<<<dim3(DLL / 64, M / 128), 256, 0, stream>>>(xbf, wdT, lat, M, DLL, DD);
  // 4) layernorm(DL) -> bf16
  mla_ln256_kernel<<<M, 256, 0, stream>>>(lat, ln_g, ln_b, latbf);
  // 5) kv = latent @ Wkv (4096x2048x256)
  mla_gemm_bf16_kernel<<<dim3(2 * DD / 64, M / 128), 256, 0, stream>>>(latbf, wkvT, kvf, M, 2 * DD, DLL);
  // 6) RoPE + head layernorm; q/k -> [b,h,s,dh], v -> [b,h,dh,s]
  mla_ropeln_kernel<<<M * HH, 64, 0, stream>>>(qpre, kvf, pos, qn_g, qn_b, kn_g, kn_b,
                                               qbf, kbf, vtbf);
  // 7) flash attention
  mla_attn_kernel<<<dim3(SS / 64, BS * HH), 256, 0, stream>>>(qbf, kbf, vtbf, attnbf);
  // 8) out = attn @ Wo (4096x1024x1024), fp32 output
  mla_gemm_bf16_kernel<<<dim3(DD / 64, M / 128), 256, 0, stream>>>(attnbf, woT, out, M, DD, DD);
}
